// MultiHeadAttention_52235392254074
// MI455X (gfx1250) — matrix operand order repaired and checked by reading, not compiled, not run on hardware
//
#include <hip/hip_runtime.h>
#include <hip/hip_bf16.h>

#define DIM    1024
#define NHEAD  16
#define HDIM   64
#define BATCH  2
#define SEQ    2048

typedef __attribute__((ext_vector_type(16))) __bf16 bf16x16;
typedef __attribute__((ext_vector_type(8)))  float  floatx8;
typedef __attribute__((ext_vector_type(4)))  unsigned u32x4;
typedef __attribute__((ext_vector_type(8)))  int      i32x8;
typedef __attribute__((ext_vector_type(4)))  int      i32x4;

union BF16Frag {
    bf16x16 v;
    uint4   q[2];
};

__device__ __forceinline__ floatx8 zero8() {
    floatx8 z;
#pragma unroll
    for (int i = 0; i < 8; ++i) z[i] = 0.0f;
    return z;
}

// ---------------------------------------------------------------------------
// DPP ROW_XMASK lane swap within 16-lane rows (gfx12 DPP16 ctrl 0x160|mask)
// ---------------------------------------------------------------------------
template <int MASK>
__device__ __forceinline__ float dpp_xor16(float x) {
    int xi = __builtin_bit_cast(int, x);
    int r  = __builtin_amdgcn_update_dpp(xi, xi, 0x160 | MASK, 0xf, 0xf, true);
    return __builtin_bit_cast(float, r);
}

__device__ __forceinline__ float rowmax16(float x) {
    x = fmaxf(x, dpp_xor16<1>(x));
    x = fmaxf(x, dpp_xor16<2>(x));
    x = fmaxf(x, dpp_xor16<4>(x));
    x = fmaxf(x, dpp_xor16<8>(x));
    return x;
}

__device__ __forceinline__ float rowsum16(float x) {
    x += dpp_xor16<1>(x);
    x += dpp_xor16<2>(x);
    x += dpp_xor16<4>(x);
    x += dpp_xor16<8>(x);
    return x;
}

// ---------------------------------------------------------------------------
// f32 -> bf16 elementwise convert
// ---------------------------------------------------------------------------
__global__ void cvt_f32_to_bf16(const float* __restrict__ in,
                                __bf16* __restrict__ out, int n) {
    int i = blockIdx.x * blockDim.x + threadIdx.x;
    if (i < n) out[i] = (__bf16)in[i];
}

// ---------------------------------------------------------------------------
// f32 [K][N] -> bf16 [N][K] transpose-convert
// ---------------------------------------------------------------------------
__global__ void transpose_cvt_bf16(const float* __restrict__ in,
                                   __bf16* __restrict__ out, int K, int N) {
    int i = blockIdx.x * blockDim.x + threadIdx.x;
    if (i < K * N) {
        int k = i / N;
        int n = i - k * N;
        out[(size_t)n * K + k] = (__bf16)in[i];
    }
}

// ---------------------------------------------------------------------------
// WMMA GEMM: C[M][N] = A[M][K](bf16, row-major) * Bt[N][K](bf16)  (+ bias)
// Block: 256 threads = 8 waves; wave computes 16 rows x 64 cols.
// ---------------------------------------------------------------------------
template <bool BF16_OUT>
__global__ void __launch_bounds__(256) gemm_bf16_wmma(
    const __bf16* __restrict__ A,
    const __bf16* __restrict__ Bt,
    const float*  __restrict__ bias,
    void* __restrict__ Cout,
    int M, int N, int K)
{
    const int lane = threadIdx.x & 31;
    const int wave = threadIdx.x >> 5;
    const int half = lane >> 4;
    const int l16  = lane & 15;
    const int m0   = blockIdx.y * 128 + wave * 16;
    const int n0   = blockIdx.x * 64;

    floatx8 acc[4];
#pragma unroll
    for (int t = 0; t < 4; ++t) acc[t] = zero8();

    const __bf16* arow = A + (size_t)(m0 + l16) * K;

    for (int k0 = 0; k0 < K; k0 += 32) {
        if (k0 + 32 < K) __builtin_prefetch(arow + k0 + 32, 0, 3);

        BF16Frag a;
        a.q[0] = *(const uint4*)(arow + k0 + 8 * half);
        a.q[1] = *(const uint4*)(arow + k0 + 16 + 8 * half);
#pragma unroll
        for (int t = 0; t < 4; ++t) {
            const __bf16* bcol = Bt + (size_t)(n0 + 16 * t + l16) * K + k0 + 8 * half;
            BF16Frag b;
            b.q[0] = *(const uint4*)(bcol);
            b.q[1] = *(const uint4*)(bcol + 16);
            acc[t] = __builtin_amdgcn_wmma_f32_16x16x32_bf16(
                false, a.v, false, b.v, (short)0, acc[t], false, false);
        }
    }

#pragma unroll
    for (int t = 0; t < 4; ++t) {
#pragma unroll
        for (int i = 0; i < 8; ++i) {
            int row = m0 + i + 8 * half;
            int col = n0 + 16 * t + l16;
            float v = acc[t][i];
            if (BF16_OUT) {
                ((__bf16*)Cout)[(size_t)row * N + col] = (__bf16)v;
            } else {
                ((float*)Cout)[(size_t)row * N + col] = v + bias[col];
            }
        }
    }
}

// ---------------------------------------------------------------------------
// Flash attention over bf16 qkv [B*SEQ][3*DIM] -> bf16 out [B*SEQ][DIM]
// Block: 128 threads = 4 waves; wave owns 16 query rows; 32-key tiles.
//   K tile: TDM tensor_load_to_lds (TENSORcnt), HW-padded to 72-elem rows
//   V tile: async-to-LDS raw (ASYNCcnt), then in-LDS transpose
// ---------------------------------------------------------------------------
__global__ void __launch_bounds__(128) flash_attn_bf16(
    const __bf16* __restrict__ qkv,
    __bf16* __restrict__ outc)
{
    __shared__ __bf16 lds_k[32][72];         // K tile [key][dim], padded rows
    __shared__ __bf16 lds_vraw[32][72];      // V tile raw [key][dim], padded
    __shared__ __bf16 lds_vt[64][40];        // V tile transposed [dim][key]
    __shared__ __bf16 lds_p[4][16][40];      // per-wave P re-swizzle tile

    const int lane = threadIdx.x & 31;
    const int wave = threadIdx.x >> 5;
    const int half = lane >> 4;
    const int l16  = lane & 15;
    const int bh   = blockIdx.y;
    const int b    = bh >> 4;
    const int h    = bh & (NHEAD - 1);
    const int q0   = blockIdx.x * 64 + wave * 16;

    const size_t rs = 3 * DIM;                   // row stride of qkv
    const __bf16* base = qkv + (size_t)(b * SEQ) * rs;

    // Q fragments: 16x64 = two 16x32 A fragments, loaded once
    BF16Frag aq[2];
#pragma unroll
    for (int f = 0; f < 2; ++f) {
        const __bf16* p = base + (size_t)(q0 + l16) * rs + h * HDIM + f * 32;
        aq[f].q[0] = *(const uint4*)(p + 8 * half);
        aq[f].q[1] = *(const uint4*)(p + 16 + 8 * half);
    }

#if __has_builtin(__builtin_amdgcn_tensor_load_to_lds)
    // TDM descriptor, loop-invariant parts.
    // group1: data_size=1(2B) | pad_enable | pad_interval=4 (32 DW) |
    //         pad_amount=3 (4 DW)  -> LDS rows padded 128B -> 144B (72 elems)
    i32x8 g1;
    g1[0] = (int)((1u << 16) | (1u << 20) | (4u << 22) | (3u << 25));
    g1[1] = (int)(64u << 16);      // tensor_dim0 = 64 (elems per key row)
    g1[2] = (int)(32u << 16);      // tensor_dim1 = 32 (keys)
    g1[3] = (int)(64u << 16);      // tile_dim0 = 64
    g1[4] = (int)32u;              // tile_dim1 = 32, tile_dim2 = 0
    g1[5] = (int)(3 * DIM);        // tensor_dim0_stride = 3072 elems
    g1[6] = 0;
    g1[7] = 0;
    const unsigned ldsk_addr = (unsigned)(uintptr_t)&lds_k[0][0];
#endif

    floatx8 o[4];
#pragma unroll
    for (int t = 0; t < 4; ++t) o[t] = zero8();
    float mrow[8], lrow[8];
#pragma unroll
    for (int i = 0; i < 8; ++i) { mrow[i] = -1.0e30f; lrow[i] = 0.0f; }

    const float SCL = 0.125f * 1.44269504088896340736f;  // 1/sqrt(d) * log2(e)

    for (int kb = 0; kb < SEQ; kb += 32) {
        const __bf16* kbase = base + (size_t)kb * rs + DIM + h * HDIM;
        const __bf16* vbase = base + (size_t)kb * rs + 2 * DIM + h * HDIM;

        // --- K tile (32x64, 4 KB) via Tensor Data Mover ---------------------
        if (wave == 0) {
#if __has_builtin(__builtin_amdgcn_tensor_load_to_lds)
            unsigned long long ga = (unsigned long long)(uintptr_t)kbase;
            u32x4 g0;
            g0[0] = 1u;                               // count=1, user mode
            g0[1] = ldsk_addr;                        // LDS dest (bytes)
            g0[2] = (unsigned)ga;                     // global_addr[31:0]
            g0[3] = (unsigned)((ga >> 32) & 0x1FFFFFFull) | (2u << 30); // type=2
#if __clang_major__ >= 23
            __builtin_amdgcn_tensor_load_to_lds(g0, g1, (i32x4)0, (i32x4)0,
                                                (i32x8)0, 0);
#else
            __builtin_amdgcn_tensor_load_to_lds(g0, g1, (i32x4)0, (i32x4)0, 0);
#endif
#else
            // fallback: async-to-LDS, 256 chunks of 16B over wave 0's 32 lanes
#pragma unroll
            for (int r = 0; r < 8; ++r) {
                int idx = lane + r * 32;
                int key = idx >> 3;
                int oct = idx & 7;
                unsigned ldsoff = (unsigned)(uintptr_t)&lds_k[key][oct * 8];
                unsigned long long gak =
                    (unsigned long long)(uintptr_t)(kbase + (size_t)key * rs + oct * 8);
                asm volatile("global_load_async_to_lds_b128 %0, %1, off"
                             :: "v"(ldsoff), "v"(gak) : "memory");
            }
#endif
        }

        // --- V tile raw (32x64) via async-to-LDS, all 128 threads -----------
#pragma unroll
        for (int r = 0; r < 2; ++r) {
            int idx = threadIdx.x + r * 128;          // 0..255
            int key = idx >> 3;
            int oct = idx & 7;                        // 8 bf16 per 16B chunk
            unsigned ldsoff = (unsigned)(uintptr_t)&lds_vraw[key][oct * 8];
            unsigned long long gav =
                (unsigned long long)(uintptr_t)(vbase + (size_t)key * rs + oct * 8);
            asm volatile("global_load_async_to_lds_b128 %0, %1, off"
                         :: "v"(ldsoff), "v"(gav) : "memory");
        }

        asm volatile("s_wait_asynccnt 0x0" ::: "memory");
#if __has_builtin(__builtin_amdgcn_tensor_load_to_lds)
        if (wave == 0) __builtin_amdgcn_s_wait_tensorcnt(0);
#endif
        __syncthreads();

        // --- in-LDS transpose: lds_vraw[key][dim] -> lds_vt[dim][key] -------
#pragma unroll
        for (int it = 0; it < 16; ++it) {
            int idx = threadIdx.x + it * 128;    // 0..2047
            int dd  = idx & 63;
            int j2  = idx >> 6;
            lds_vt[dd][j2] = lds_vraw[j2][dd];
        }
        __syncthreads();

        // --- S = Q * K^T for two 16-key subtiles; B frags from LDS K tile ---
        floatx8 s0 = zero8(), s1 = zero8();
#pragma unroll
        for (int f = 0; f < 2; ++f) {
            const __bf16* k0p = &lds_k[l16][f * 32 + 8 * half];
            const __bf16* k1p = &lds_k[16 + l16][f * 32 + 8 * half];
            BF16Frag b0, b1;
            b0.q[0] = *(const uint4*)(k0p);
            b0.q[1] = *(const uint4*)(k0p + 16);
            b1.q[0] = *(const uint4*)(k1p);
            b1.q[1] = *(const uint4*)(k1p + 16);
            s0 = __builtin_amdgcn_wmma_f32_16x16x32_bf16(
                false, aq[f].v, false, b0.v, (short)0, s0, false, false);
            s1 = __builtin_amdgcn_wmma_f32_16x16x32_bf16(
                false, aq[f].v, false, b1.v, (short)0, s1, false, false);
        }

        // --- online softmax (exp2 domain, DPP ROW_XMASK row reductions) -----
#pragma unroll
        for (int i = 0; i < 8; ++i) {
            float x0 = s0[i] * SCL;
            float x1 = s1[i] * SCL;
            float mx    = rowmax16(fmaxf(x0, x1));
            float mnew  = fmaxf(mrow[i], mx);
            float alpha = exp2f(mrow[i] - mnew);
            mrow[i] = mnew;
            float p0 = exp2f(x0 - mnew);
            float p1 = exp2f(x1 - mnew);
            float rsum = rowsum16(p0 + p1);
            lrow[i] = lrow[i] * alpha + rsum;
#pragma unroll
            for (int t = 0; t < 4; ++t) o[t][i] *= alpha;
            int prow = i + 8 * half;
            lds_p[wave][prow][l16]      = (__bf16)p0;
            lds_p[wave][prow][16 + l16] = (__bf16)p1;
        }
        asm volatile("s_wait_dscnt 0x0" ::: "memory");

        // re-read P as an A fragment (16x32)
        BF16Frag pa;
        {
            const __bf16* pp = &lds_p[wave][l16][0];
            pa.q[0] = *(const uint4*)(pp + 8 * half);
            pa.q[1] = *(const uint4*)(pp + 16 + 8 * half);
        }

        // --- O += P * V -----------------------------------------------------
#pragma unroll
        for (int t = 0; t < 4; ++t) {
            const __bf16* vp = &lds_vt[16 * t + l16][8 * half];
            BF16Frag bv;
            bv.q[0] = *(const uint4*)(vp);
            bv.q[1] = *(const uint4*)(vp + 16);
            o[t] = __builtin_amdgcn_wmma_f32_16x16x32_bf16(
                false, pa.v, false, bv.v, (short)0, o[t], false, false);
        }
        __syncthreads();
    }

    // normalize and store head-concatenated bf16 output
    __bf16* orow = outc + (size_t)(b * SEQ) * DIM + h * HDIM;
#pragma unroll
    for (int i = 0; i < 8; ++i) {
        float inv = 1.0f / lrow[i];
        int row = q0 + i + 8 * half;
#pragma unroll
        for (int t = 0; t < 4; ++t) {
            orow[(size_t)row * DIM + 16 * t + l16] = (__bf16)(o[t][i] * inv);
        }
    }
}

// ---------------------------------------------------------------------------
// Launch
// ---------------------------------------------------------------------------
extern "C" void kernel_launch(void* const* d_in, const int* in_sizes, int n_in,
                              void* d_out, int out_size, void* d_ws, size_t ws_size,
                              hipStream_t stream) {
    const float* x      = (const float*)d_in[0];   // [2,2048,1024]
    const float* W_qkv  = (const float*)d_in[1];   // [1024,3072]
    const float* W_proj = (const float*)d_in[2];   // [1024,1024]
    const float* b_proj = (const float*)d_in[3];   // [1024]
    float* out = (float*)d_out;                    // [2,2048,1024]

    char* ws = (char*)d_ws;
    __bf16* xb      = (__bf16*)(ws);                          //  8 MB: x bf16
    __bf16* wqkv_t  = (__bf16*)(ws + (size_t)(8u  << 20));    //  6 MB: W_qkv^T
    __bf16* wproj_t = (__bf16*)(ws + (size_t)(14u << 20));    //  2 MB: W_proj^T
    __bf16* qkv_b   = (__bf16*)(ws + (size_t)(16u << 20));    // 24 MB: qkv bf16
    __bf16* attn_b  = (__bf16*)(ws + (size_t)(40u << 20));    //  8 MB: attn out

    const int M = BATCH * SEQ;                     // 4096

    {
        int n = M * DIM;
        cvt_f32_to_bf16<<<(n + 255) / 256, 256, 0, stream>>>(x, xb, n);
    }
    {
        int n = DIM * 3 * DIM;
        transpose_cvt_bf16<<<(n + 255) / 256, 256, 0, stream>>>(W_qkv, wqkv_t, DIM, 3 * DIM);
    }
    {
        int n = DIM * DIM;
        transpose_cvt_bf16<<<(n + 255) / 256, 256, 0, stream>>>(W_proj, wproj_t, DIM, DIM);
    }

    gemm_bf16_wmma<true><<<dim3((3 * DIM) / 64, M / 128), 256, 0, stream>>>(
        xb, wqkv_t, nullptr, qkv_b, M, 3 * DIM, DIM);

    flash_attn_bf16<<<dim3(SEQ / 64, BATCH * NHEAD), 128, 0, stream>>>(qkv_b, attn_b);

    gemm_bf16_wmma<false><<<dim3(DIM / 64, M / 128), 256, 0, stream>>>(
        attn_b, wproj_t, b_proj, out, M, DIM, DIM);
}
